// Net_51032801411073
// MI455X (gfx1250) — matrix operand order repaired and checked by reading, not compiled, not run on hardware
//
#include <hip/hip_runtime.h>

#define N_NODES 50000
#define N_EDGES 800000
#define F_IN    128
#define HID     32
#define HEADS   4
#define NCLS    16
#define D1      128   /* HEADS*HID */
#define TOPK    8
#define NEG_SLOPE 0.2f
#define MT      5     /* m-tiles (16 rows each) per block in the GEMM */

typedef __attribute__((ext_vector_type(16))) _Float16 v16h;
typedef __attribute__((ext_vector_type(8)))  _Float16 v8h;
typedef __attribute__((ext_vector_type(8)))  float    v8f;

// ---- Tensor Data Mover availability / arity probing --------------------
#if defined(__has_builtin)
#  if __has_builtin(__builtin_amdgcn_tensor_load_to_lds)
#    define HAVE_TDM 1
#  endif
#endif
#if __has_include(<hip/amd_detail/amd_gfx1250_TDM.h>)
#  define TDM_ARGS6 1   /* amdgpu-toolchain (clang-23): 6-arg builtin */
#endif

#ifdef HAVE_TDM
typedef unsigned int v4u __attribute__((ext_vector_type(4)));
typedef int          v4i __attribute__((ext_vector_type(4)));
typedef int          v8i __attribute__((ext_vector_type(8)));

// 1-D TDM tile load: n_elems contiguous f16 from gsrc -> LDS byte offset lds_off.
// D# per CDNA5 ISA 8.3/8.4: group0 = {count=1, lds_addr, global_addr[56:0], type=2},
// group1 = {wg_mask=0, data_size=1(2B), tensor_dim0=tile_dim0=n_elems, dim1 unused}.
__device__ __forceinline__ void tdm_load_1d(const _Float16* gsrc, unsigned lds_off,
                                            unsigned n_elems) {
  unsigned long long ga = (unsigned long long)gsrc;
  v4u g0;
  g0[0] = 1u;                                                  // count=1 (valid)
  g0[1] = lds_off;                                             // lds_addr (bytes)
  g0[2] = (unsigned)(ga & 0xFFFFFFFFu);                        // global_addr[31:0]
  g0[3] = (unsigned)((ga >> 32) & 0x01FFFFFFu) | 0x80000000u;  // addr[56:32] | type=2
  v8i g1;
  g1[0] = (int)(1u << 16);                       // data_size=1 (2 bytes); wg_mask=0
  g1[1] = (int)((n_elems & 0xFFFFu) << 16);      // tensor_dim0[15:0]  (bits 63:48)
  g1[2] = (int)(((n_elems >> 16) & 0xFFFFu)      // tensor_dim0[31:16] (bits 79:64)
                | (1u << 16));                   // tensor_dim1 = 1    (bits 95:80)
  g1[3] = (int)(n_elems << 16);                  // tile_dim0          (bits 127:112)
  g1[4] = 0;                                     // tile_dim1 = tile_dim2 = 0 (1-D)
  g1[5] = (int)n_elems;                          // tensor_dim0_stride[31:0]
  g1[6] = 0;
  g1[7] = 0;
  v4i g2 = {0, 0, 0, 0}, g3 = {0, 0, 0, 0};
#ifdef TDM_ARGS6
  v8i g4 = {0, 0, 0, 0, 0, 0, 0, 0};
  __builtin_amdgcn_tensor_load_to_lds(g0, g1, g2, g3, g4, 0);
#else
  __builtin_amdgcn_tensor_load_to_lds(g0, g1, g2, g3, 0);
#endif
}
#endif // HAVE_TDM

// ---------------------------------------------------------------- utilities
__device__ __forceinline__ unsigned encf(float x) {
  unsigned u = __float_as_uint(x);
  return (u & 0x80000000u) ? ~u : (u | 0x80000000u);
}

__global__ void k_fill_u32(unsigned* p, unsigned v, long n) {
  long i = (long)blockIdx.x * blockDim.x + threadIdx.x;
  if (i < n) p[i] = v;
}

__global__ void k_cvt_f16(_Float16* dst, const float* src, long n) {
  long i = (long)blockIdx.x * blockDim.x + threadIdx.x;
  if (i < n) dst[i] = (_Float16)src[i];
}

// dst[N][K] (f16) = transpose of src[K][N] (f32)
__global__ void k_transpose_cvt(_Float16* dst, const float* src, int K, int N) {
  long i = (long)blockIdx.x * blockDim.x + threadIdx.x;
  if (i >= (long)K * N) return;
  int k = (int)(i / N), n = (int)(i % N);
  dst[(long)n * K + k] = (_Float16)src[i];
}

__global__ void k_bias_elu_f16(const float* in, const float* b, _Float16* o, long n, int F) {
  long i = (long)blockIdx.x * blockDim.x + threadIdx.x;
  if (i >= n) return;
  float v = in[i] + b[i % F];
  v = v > 0.f ? v : (expf(v) - 1.f);
  o[i] = (_Float16)v;
}

__global__ void k_bias_f16(const float* in, const float* b, _Float16* o, long n, int F) {
  long i = (long)blockIdx.x * blockDim.x + threadIdx.x;
  if (i >= n) return;
  o[i] = (_Float16)(in[i] + b[i % F]);
}

__global__ void k_log_softmax16(const float* in, const float* bias, float* out, int rows) {
  int r = blockIdx.x * blockDim.x + threadIdx.x;
  if (r >= rows) return;
  float v[16]; float mx = -__builtin_inff();
  #pragma unroll
  for (int i = 0; i < 16; ++i) { v[i] = in[(long)r * 16 + i] + bias[i]; mx = fmaxf(mx, v[i]); }
  float s = 0.f;
  #pragma unroll
  for (int i = 0; i < 16; ++i) s += expf(v[i] - mx);
  float l = logf(s);
  #pragma unroll
  for (int i = 0; i < 16; ++i) out[(long)r * 16 + i] = v[i] - mx - l;
}

// -------------------------------------------------------------- WMMA GEMM
// C[M,N](f32) = A[M,128](f16 row-major) * BT[N,128](f16 row-major).  N is a
// template constant (128 or 16) so C stores use immediate offsets.
// Block = 4 waves; wave <-> 16-col n-tile; block walks MT m-tiles of 16 rows.
// B fragments (4 x v16h) are loaded ONCE per wave and stay register-resident.
// TDM path: A tile (16x128 f16 = 4KB, contiguous) is DMA'd into double-buffered
// LDS via tensor_load_to_lds; s_wait_tensorcnt + barrier hand-off overlaps the
// DMA of tile t+1 with WMMA on tile t.  The asm escape of &sA forces the
// compiler to treat the array as DMA-written (keeps ds_load_b128 alive).
template <int N>
__global__ void k_wmma_gemm(const _Float16* __restrict__ A,
                            const _Float16* __restrict__ BT,
                            float* __restrict__ C, int M) {
  const int K = 128;
  int wave = threadIdx.x >> 5;
  int lane = threadIdx.x & 31;
  int half = lane >> 4, l15 = lane & 15;
  int n0 = (blockIdx.y * 4 + wave) * 16;
  bool active = (n0 < N);
  int mt0 = blockIdx.x * MT;
  if (mt0 * 16 >= M) return;   // block-uniform

  // Preload all B fragments (K=128 -> 4 k-steps); ISA 7.12.2 B 32x16 layout:
  // lanes 0-15 col N=l hold K=0..7,16..23, lanes 16-31 hold K=8..15,24..31 (BT row-contig).
  v16h bf[4];
  if (active) {
    const _Float16* brow = BT + (long)(n0 + l15) * K;
    #pragma unroll
    for (int ks = 0; ks < 4; ++ks) {
      v8h blo = *(const v8h*)(brow + ks * 32 + half * 8);
      v8h bhi = *(const v8h*)(brow + ks * 32 + 16 + half * 8);
      #pragma unroll
      for (int i = 0; i < 8; ++i) { bf[ks][i] = blo[i]; bf[ks][i + 8] = bhi[i]; }
    }
  }

#ifdef HAVE_TDM
  __shared__ __align__(16) _Float16 sA[2][16 * 128];   // double-buffered A tile
  _Float16* sap = &sA[0][0];
  if (wave == 0)
    tdm_load_1d(A + (long)mt0 * 16 * K, 0u, 16 * K);
  for (int t = 0; t < MT; ++t) {
    int mtile = mt0 + t;
    if (mtile * 16 >= M) break;                        // block-uniform
    if (wave == 0) __builtin_amdgcn_s_wait_tensorcnt(0);
    __syncthreads();                                   // tile t visible to all
    // DMA wrote sA behind the compiler's back: escape + memory clobber.
    asm volatile("" :: "v"(sap) : "memory");
    if (wave == 0 && t + 1 < MT && (mtile + 1) * 16 < M)
      tdm_load_1d(A + (long)(mtile + 1) * 16 * K,
                  (unsigned)(((t + 1) & 1) * 16 * K * 2), 16 * K);
    if (active) {
      const _Float16* arow = &sA[t & 1][l15 * K];
      v8f acc = {};
      #pragma unroll
      for (int ks = 0; ks < 4; ++ks) {
        v8h alo = *(const v8h*)(arow + ks * 32 + half * 8);       // ds_load_b128
        v8h ahi = *(const v8h*)(arow + ks * 32 + 16 + half * 8);
        v16h a;
        #pragma unroll
        for (int i = 0; i < 8; ++i) { a[i] = alo[i]; a[i + 8] = ahi[i]; }
        acc = __builtin_amdgcn_wmma_f32_16x16x32_f16(false, a, false, bf[ks],
                                                     (short)0, acc, false, false);
      }
      float* crow = C + (long)(mtile * 16 + half * 8) * N + n0 + l15;
      #pragma unroll
      for (int j = 0; j < 8; ++j) crow[j * N] = acc[j];   // immediate offsets
    }
    __syncthreads();                                   // buf reusable next iter
  }
#else
  for (int t = 0; t < MT; ++t) {
    int mtile = mt0 + t;
    if (mtile * 16 >= M) break;
    if (!active) continue;
    const _Float16* arow = A + (long)(mtile * 16 + l15) * K;
    v8f acc = {};
    #pragma unroll
    for (int ks = 0; ks < 4; ++ks) {
      v8h alo = *(const v8h*)(arow + ks * 32 + half * 8);
      v8h ahi = *(const v8h*)(arow + ks * 32 + 16 + half * 8);
      v16h a;
      #pragma unroll
      for (int i = 0; i < 8; ++i) { a[i] = alo[i]; a[i + 8] = ahi[i]; }
      acc = __builtin_amdgcn_wmma_f32_16x16x32_f16(false, a, false, bf[ks],
                                                   (short)0, acc, false, false);
    }
    float* crow = C + (long)(mtile * 16 + half * 8) * N + n0 + l15;
    #pragma unroll
    for (int j = 0; j < 8; ++j) crow[j * N] = acc[j];
  }
#endif
}

// ------------------------------------------------------------ edge kernels
// One wave per (edge, head); lane = channel.  score = sum_c lrelu(xl+xr)*att
__global__ void k_edge_score(const float* __restrict__ xl, const float* __restrict__ xr,
                             const long long* __restrict__ srcI, const long long* __restrict__ dstI,
                             const float* __restrict__ att, float* __restrict__ score,
                             int E, int H, int C) {
  int wpb = blockDim.x >> 5;
  long pair = (long)blockIdx.x * wpb + (threadIdx.x >> 5);
  int lane = threadIdx.x & 31;
  if (pair >= (long)E * H) return;
  int e = (int)(pair / H), h = (int)(pair % H);
  long s = (long)srcI[e], d = (long)dstI[e];
  float acc = 0.f;
  for (int c = lane; c < C; c += 32) {
    float v = xl[s * H * C + h * C + c] + xr[d * H * C + h * C + c];
    v = v > 0.f ? v : NEG_SLOPE * v;
    acc += v * att[h * C + c];
  }
  #pragma unroll
  for (int off = 16; off; off >>= 1) acc += __shfl_xor(acc, off, 32);
  if (lane == 0) score[pair] = acc;
}

// Round r: per (dst,head) max of scores strictly below previous round's max.
__global__ void k_thr_round(const float* __restrict__ score, const long long* __restrict__ dstI,
                            const unsigned* __restrict__ prev, unsigned* __restrict__ cur,
                            int E, int H, int first) {
  long i = (long)blockIdx.x * blockDim.x + threadIdx.x;
  if (i >= (long)E * H) return;
  int e = (int)(i / H), h = (int)(i % H);
  long d = (long)dstI[e] * H + h;
  unsigned es = encf(score[i]);
  unsigned ep = first ? 0xFFFFFFFFu : prev[d];
  if (es < ep) atomicMax(&cur[d], es);
}

// kthmode=0: empty -> 0.0 (reference's isfinite guard);  kthmode=1: empty -> -inf
__global__ void k_decode(const unsigned* enc, float* out, long n, int kthmode) {
  long i = (long)blockIdx.x * blockDim.x + threadIdx.x;
  if (i >= n) return;
  unsigned e = enc[i];
  float v;
  if (e == 0u) v = kthmode ? -__builtin_inff() : 0.0f;
  else {
    unsigned u = (e & 0x80000000u) ? (e & 0x7FFFFFFFu) : ~e;
    v = __uint_as_float(u);
  }
  out[i] = v;
}

__global__ void k_exp_sums(const float* __restrict__ score, const long long* __restrict__ dstI,
                           const float* __restrict__ kth, const float* __restrict__ mk,
                           const float* __restrict__ mn,
                           float* __restrict__ exk, float* __restrict__ exn,
                           float* __restrict__ sumk, float* __restrict__ sumn, int E, int H) {
  long i = (long)blockIdx.x * blockDim.x + threadIdx.x;
  if (i >= (long)E * H) return;
  int e = (int)(i / H), h = (int)(i % H);
  long d = (long)dstI[e] * H + h;
  float s = score[i];
  float ek = 0.f, en = 0.f;
  if (s >= kth[d]) ek = expf(s - mk[d]);
  else             en = expf(s - mn[d]);
  exk[i] = ek; exn[i] = en;
  if (ek != 0.f) atomicAdd(&sumk[d], ek);
  if (en != 0.f) atomicAdd(&sumn[d], en);
}

// One wave per (edge,head); lane = channel.  out += alpha*xl[src], noise += alpha_n*xl[src]
__global__ void k_aggregate(const float* __restrict__ xl,
                            const long long* __restrict__ srcI, const long long* __restrict__ dstI,
                            const float* __restrict__ exk, const float* __restrict__ exn,
                            const float* __restrict__ sumk, const float* __restrict__ sumn,
                            float* __restrict__ out, float* __restrict__ noise,
                            int E, int H, int C) {
  int wpb = blockDim.x >> 5;
  long pair = (long)blockIdx.x * wpb + (threadIdx.x >> 5);
  int lane = threadIdx.x & 31;
  if (pair >= (long)E * H) return;
  int e = (int)(pair / H), h = (int)(pair % H);
  long s = (long)srcI[e], d = (long)dstI[e];
  long seg = d * H + h;
  float ak = exk[pair] / (sumk[seg] + 1e-16f);
  float an = exn[pair] / (sumn[seg] + 1e-16f);
  if (lane < C) {
    float m = xl[s * H * C + h * C + lane];
    if (ak != 0.f) atomicAdd(&out[d * H * C + h * C + lane], ak * m);
    if (an != 0.f) atomicAdd(&noise[d * H * C + h * C + lane], an * m);
  }
}

// ------------------------------------------------------------------ host
static void launch_gemm(const _Float16* A, const _Float16* BT, float* C,
                        int M, int N, hipStream_t s) {
  dim3 block(128);
  dim3 grid((M / 16 + MT - 1) / MT, (N + 63) / 64);
  if (N == 128)     k_wmma_gemm<128><<<grid, block, 0, s>>>(A, BT, C, M);
  else /* N==16 */  k_wmma_gemm<16><<<grid, block, 0, s>>>(A, BT, C, M);
}

extern "C" void kernel_launch(void* const* d_in, const int* in_sizes, int n_in,
                              void* d_out, int out_size, void* d_ws, size_t ws_size,
                              hipStream_t stream) {
  (void)in_sizes; (void)n_in; (void)out_size; (void)ws_size;
  const float*      x    = (const float*)d_in[0];
  const long long*  ei   = (const long long*)d_in[1];   // int64 [2,E]
  const long long*  srcI = ei;
  const long long*  dstI = ei + N_EDGES;
  // d_in[2] = attn_topk (known == 8, compiled in)
  const float* Wl1  = (const float*)d_in[3];
  const float* Wr1  = (const float*)d_in[4];
  const float* att1 = (const float*)d_in[5];
  const float* b1   = (const float*)d_in[6];
  const float* Wl2  = (const float*)d_in[7];
  const float* Wr2  = (const float*)d_in[8];
  const float* att2 = (const float*)d_in[9];
  const float* b2   = (const float*)d_in[10];
  const float* l1w  = (const float*)d_in[11];
  const float* l1b  = (const float*)d_in[12];
  const float* l2w  = (const float*)d_in[13];
  const float* l2b  = (const float*)d_in[14];
  float* out = (float*)d_out;

  // ---- workspace bump allocator ----
  char* w = (char*)d_ws; size_t off = 0;
  auto alloc = [&](size_t bytes) -> void* {
    void* p = w + off; off = (off + bytes + 255) & ~(size_t)255; return p;
  };
  const long NF = (long)N_NODES * D1;          // 6.4M elems
  _Float16* x_h   = (_Float16*)alloc(NF * 2);  // reused later as t1h
  _Float16* wl1t  = (_Float16*)alloc(128 * 128 * 2);
  _Float16* wr1t  = (_Float16*)alloc(128 * 128 * 2);
  _Float16* lin1t = (_Float16*)alloc(128 * 128 * 2);
  _Float16* wl2t  = (_Float16*)alloc(16 * 128 * 2);
  _Float16* wr2t  = (_Float16*)alloc(16 * 128 * 2);
  _Float16* lin2t = (_Float16*)alloc(16 * 128 * 2);
  float* xl1    = (float*)alloc(NF * 4);                 // reused as t1
  float* xr1    = (float*)alloc(NF * 4);                 // carved for layer-2 buffers
  float* score1 = (float*)alloc((long)N_EDGES * HEADS * 4); // reused as exk2/exn2
  unsigned* thrA = (unsigned*)alloc((long)N_NODES * HEADS * 4);
  unsigned* thrB = (unsigned*)alloc((long)N_NODES * HEADS * 4);
  float* mk   = (float*)alloc((long)N_NODES * HEADS * 4);
  float* kth  = (float*)alloc((long)N_NODES * HEADS * 4);
  float* mn   = (float*)alloc((long)N_NODES * HEADS * 4);
  float* sumk = (float*)alloc((long)N_NODES * HEADS * 4);
  float* sumn = (float*)alloc((long)N_NODES * HEADS * 4);
  float* exk1 = (float*)alloc((long)N_EDGES * HEADS * 4);
  float* exn1 = (float*)alloc((long)N_EDGES * HEADS * 4);
  float* out1   = (float*)alloc(NF * 4);
  float* noise1 = (float*)alloc(NF * 4);
  _Float16* h_h  = (_Float16*)alloc(NF * 2);
  _Float16* nb_h = (_Float16*)alloc(NF * 2);
  // aliases (lifetimes disjoint):
  float*    t1  = xl1;
  _Float16* t1h = x_h;
  float* xl2    = xr1;                    // each piece = 50000*16 floats
  float* xr2    = xr1 + 800000;
  float* t2     = xr1 + 1600000;
  float* out2   = xr1 + 2400000;
  float* noise2 = xr1 + 3200000;
  float* score2 = xr1 + 4000000;          // E floats
  float* exk2   = score1;
  float* exn2   = score1 + N_EDGES;

  auto gb = [](long n) { return (int)((n + 255) / 256); };

  // ---- 9-round segment top-k thresholds (shared for both layers) ----
  auto thr_rounds = [&](const float* score, int H, float* mkO, float* kthO, float* mnO) {
    long NH = (long)N_NODES * H, EH = (long)N_EDGES * H;
    for (int r = 0; r <= 8; ++r) {
      unsigned* cur  = (r & 1) ? thrB : thrA;
      unsigned* prev = (r & 1) ? thrA : thrB;
      k_fill_u32<<<gb(NH), 256, 0, stream>>>(cur, 0u, NH);
      k_thr_round<<<gb(EH), 256, 0, stream>>>(score, dstI, prev, cur, N_EDGES, H, r == 0);
      if (r == 0) k_decode<<<gb(NH), 256, 0, stream>>>(cur, mkO, NH, 0);
      if (r == 7) k_decode<<<gb(NH), 256, 0, stream>>>(cur, kthO, NH, 1);
      if (r == 8) k_decode<<<gb(NH), 256, 0, stream>>>(cur, mnO, NH, 0);
    }
  };
  auto edge_layer = [&](const float* xl, const float* xr, const float* att, float* score,
                        float* exk, float* exn, float* o, float* nz, int H, int C) {
    long EH = (long)N_EDGES * H, NH = (long)N_NODES * H;
    long NHC = (long)N_NODES * H * C;
    int blkPairs = (int)((EH + 7) / 8);                 // 8 waves/block
    k_edge_score<<<blkPairs, 256, 0, stream>>>(xl, xr, srcI, dstI, att, score, N_EDGES, H, C);
    thr_rounds(score, H, mk, kth, mn);
    k_fill_u32<<<gb(NH), 256, 0, stream>>>((unsigned*)sumk, 0u, NH);
    k_fill_u32<<<gb(NH), 256, 0, stream>>>((unsigned*)sumn, 0u, NH);
    k_exp_sums<<<gb(EH), 256, 0, stream>>>(score, dstI, kth, mk, mn, exk, exn, sumk, sumn,
                                           N_EDGES, H);
    k_fill_u32<<<gb(NHC), 256, 0, stream>>>((unsigned*)o, 0u, NHC);
    k_fill_u32<<<gb(NHC), 256, 0, stream>>>((unsigned*)nz, 0u, NHC);
    k_aggregate<<<blkPairs, 256, 0, stream>>>(xl, srcI, dstI, exk, exn, sumk, sumn, o, nz,
                                              N_EDGES, H, C);
  };

  // ---- prep: f16 conversions / weight transposes ----
  k_cvt_f16<<<gb(NF), 256, 0, stream>>>(x_h, x, NF);
  k_transpose_cvt<<<gb(128 * 128), 256, 0, stream>>>(wl1t, Wl1, 128, 128);
  k_transpose_cvt<<<gb(128 * 128), 256, 0, stream>>>(wr1t, Wr1, 128, 128);
  k_transpose_cvt<<<gb(128 * 128), 256, 0, stream>>>(lin1t, l1w, 128, 128);
  k_transpose_cvt<<<gb(128 * 16), 256, 0, stream>>>(wl2t, Wl2, 128, 16);
  k_transpose_cvt<<<gb(128 * 16), 256, 0, stream>>>(wr2t, Wr2, 128, 16);
  k_transpose_cvt<<<gb(128 * 16), 256, 0, stream>>>(lin2t, l2w, 128, 16);

  // ---- layer 1: projections (WMMA, K=128) + edge attention ----
  launch_gemm(x_h, wl1t, xl1, N_NODES, 128, stream);
  launch_gemm(x_h, wr1t, xr1, N_NODES, 128, stream);
  edge_layer(xl1, xr1, att1, score1, exk1, exn1, out1, noise1, HEADS, HID);

  // h = elu(out1 + b1) (f16 for layer-2 GEMMs);  nb = noise1 + b1 (f16 for MLP)
  k_bias_elu_f16<<<gb(NF), 256, 0, stream>>>(out1, b1, h_h, NF, D1);
  k_bias_f16<<<gb(NF), 256, 0, stream>>>(noise1, b1, nb_h, NF, D1);

  // ---- noise MLP: elu(nb@lin1+b) @ lin2 + b -> log_softmax (output slot 1) ----
  launch_gemm(nb_h, lin1t, t1, N_NODES, 128, stream);
  k_bias_elu_f16<<<gb(NF), 256, 0, stream>>>(t1, l1b, t1h, NF, 128);
  launch_gemm(t1h, lin2t, t2, N_NODES, 16, stream);
  k_log_softmax16<<<gb(N_NODES), 256, 0, stream>>>(t2, l2b, out + (long)N_NODES * NCLS, N_NODES);

  // ---- layer 2 (H=1, C=16) ----
  launch_gemm(h_h, wl2t, xl2, N_NODES, 16, stream);
  launch_gemm(h_h, wr2t, xr2, N_NODES, 16, stream);
  edge_layer(xl2, xr2, att2, score2, exk2, exn2, out2, noise2, 1, NCLS);

  // outputs: slot 0 = log_softmax(out2 + b2), slot 2 = log_softmax(noise2 + b2)
  k_log_softmax16<<<gb(N_NODES), 256, 0, stream>>>(out2, b2, out, N_NODES);
  k_log_softmax16<<<gb(N_NODES), 256, 0, stream>>>(noise2, b2, out + 2L * N_NODES * NCLS, N_NODES);
}
